// LSTM_66391604461886
// MI455X (gfx1250) — matrix operand order repaired and checked by reading, not compiled, not run on hardware
//
#include <hip/hip_runtime.h>
#include <cstdint>

// ---------------------------------------------------------------------------
// Persistent 2-layer LSTM for MI455X (gfx1250, wave32, WMMA).
//   B=256, T=1024, D=6, H=256.
// Batch is split into 16 independent tiles of 16 rows; one workgroup (256
// threads = 8 waves) owns a tile and runs the full T-loop. GEMMs use
// v_wmma_f32_16x16x32_bf16 with fp32 accumulators initialized to the biases.
// Weights are pre-converted to bf16 in WMMA B-fragment order in d_ws.
// ---------------------------------------------------------------------------

#define Bsz 256
#define Tn  1024
#define Dn  6
#define Hn  256

typedef __attribute__((ext_vector_type(16))) __bf16        v16bf;
typedef __attribute__((ext_vector_type(8)))  float         v8f;
typedef __attribute__((ext_vector_type(4)))  unsigned int  u32x4;

union Frag {
    u32x4          q[2];   // two 16B chunks (32 bytes = 16 bf16)
    unsigned short s[16];
    v16bf          v;
};

__device__ __forceinline__ unsigned short f2bf(float f) {
    unsigned int u = __float_as_uint(f);
    u += 0x7FFFu + ((u >> 16) & 1u);          // round-to-nearest-even
    return (unsigned short)(u >> 16);
}
__device__ __forceinline__ float bf2f(unsigned short s) {
    return __uint_as_float(((unsigned int)s) << 16);
}
__device__ __forceinline__ float sigmoidf_(float x) { return 1.f / (1.f + __expf(-x)); }
__device__ __forceinline__ float tanhf_(float x)    { return 2.f / (1.f + __expf(-2.f * x)) - 1.f; }

// -------- fragment-ordered bf16 weight layout in d_ws ----------------------
// B-fragment (K=32 x N=16) per ISA: lane l holds column n = l&15,
// k = 32*kt + 8*(l>>4) + (j<8 ? j : j+8), j = 0..15.  One fragment = 512 bf16 = 1 KB,
// stored as frag[(kt*64 + nt)*512 + l*16 + j].
// Region element offsets (bf16 units):
#define WIH1_OFF 0        //  64 frags (kt=0, K padded 6->32)
#define WHH1_OFF 32768    // 512 frags
#define WIH2_OFF 294912   // 512 frags
#define WHH2_OFF 557056   // 512 frags
#define WF_TOTAL 819200
// biases (fp32) appended after the bf16 block (byte offset 2*WF_TOTAL).

__global__ void prep_weights(const float* __restrict__ Wih1,
                             const float* __restrict__ Whh1,
                             const float* __restrict__ Wih2,
                             const float* __restrict__ Whh2,
                             const float* __restrict__ bih1,
                             const float* __restrict__ bhh1,
                             const float* __restrict__ bih2,
                             const float* __restrict__ bhh2,
                             unsigned short* __restrict__ wf,
                             float* __restrict__ biasb) {
    int idx = blockIdx.x * 256 + threadIdx.x;
    if (idx < WF_TOTAL) {
        int frag = idx >> 9;
        int off  = idx & 511;
        int l    = off >> 4;
        int j    = off & 15;
        int mrow = l & 15;
        int hi   = l >> 4;
        float v;
        if (frag < 64) {                         // W_ih1, kt = 0, K padded
            int nt = frag;
            int k  = (j < 8 ? j : j + 8) + 8 * hi;
            int n  = 16 * nt + mrow;
            v = (k < Dn) ? Wih1[n * Dn + k] : 0.f;
        } else if (frag < 576) {                 // W_hh1
            int f = frag - 64, kt = f >> 6, nt = f & 63;
            int k = 32 * kt + (j < 8 ? j : j + 8) + 8 * hi;
            int n = 16 * nt + mrow;
            v = Whh1[n * Hn + k];
        } else if (frag < 1088) {                // W_ih2
            int f = frag - 576, kt = f >> 6, nt = f & 63;
            int k = 32 * kt + (j < 8 ? j : j + 8) + 8 * hi;
            int n = 16 * nt + mrow;
            v = Wih2[n * Hn + k];
        } else {                                 // W_hh2
            int f = frag - 1088, kt = f >> 6, nt = f & 63;
            int k = 32 * kt + (j < 8 ? j : j + 8) + 8 * hi;
            int n = 16 * nt + mrow;
            v = Whh2[n * Hn + k];
        }
        wf[idx] = f2bf(v);
    } else {
        int i = idx - WF_TOTAL;
        if (i < 1024)      biasb[i] = bih1[i] + bhh1[i];
        else if (i < 2048) biasb[i] = bih2[i - 1024] + bhh2[i - 1024];
    }
}

// ---------------------------------------------------------------------------
__global__ __launch_bounds__(256) void lstm_persist(
    const float* __restrict__ x,
    const unsigned short* __restrict__ wih1,
    const unsigned short* __restrict__ whh1,
    const unsigned short* __restrict__ wih2,
    const unsigned short* __restrict__ whh2,
    const float* __restrict__ bias1,
    const float* __restrict__ bias2,
    const float* __restrict__ Wout,
    const float* __restrict__ bout,
    float* __restrict__ out) {
    __shared__ __align__(16) unsigned short h1s[16 * Hn];   // bf16 [m][k]
    __shared__ __align__(16) unsigned short h2s[16 * Hn];
    __shared__ float pr[16][16][8];                         // proj partials

    const int tid  = threadIdx.x;
    const int lane = tid & 31;
    const int wv   = tid >> 5;      // wave 0..7: owns gate columns 32*wv..32*wv+31
    const int mrow = lane & 15;     // batch row within tile (A-layout M)
    const int hi   = lane >> 4;     // lane half
    const int bb   = blockIdx.x * 16;

    for (int i = tid; i < 16 * Hn; i += 256) { h1s[i] = 0; h2s[i] = 0; }

    // warm the cache hierarchy with the recurrent weights (gfx1250 prefetch)
    for (int i = tid; i < 512; i += 256) {
        __builtin_prefetch(((const char*)whh1) + i * 1024, 0, 3);
        __builtin_prefetch(((const char*)wih2) + i * 1024, 0, 3);
        __builtin_prefetch(((const char*)whh2) + i * 1024, 0, 3);
    }

    float bs1[4][2], bs2[4][2];
#pragma unroll
    for (int g = 0; g < 4; ++g)
#pragma unroll
        for (int p = 0; p < 2; ++p) {
            int col = 32 * wv + 16 * p + mrow;
            bs1[g][p] = bias1[g * Hn + col];
            bs2[g][p] = bias2[g * Hn + col];
        }

    float c1[2][8], c2[2][8];
#pragma unroll
    for (int p = 0; p < 2; ++p)
#pragma unroll
        for (int r = 0; r < 8; ++r) { c1[p][r] = 0.f; c2[p][r] = 0.f; }

    __syncthreads();

    const long xrow = (long)(bb + mrow) * (Tn * Dn);

    for (int t = 0; t < Tn; ++t) {
        // ======================= layer 1 =======================
        v8f acc[4][2];
#pragma unroll
        for (int g = 0; g < 4; ++g)
#pragma unroll
            for (int p = 0; p < 2; ++p)
#pragma unroll
                for (int r = 0; r < 8; ++r) acc[g][p][r] = bs1[g][p];

#pragma unroll 2
        for (int kt = 0; kt < 8; ++kt) {
            Frag a;
            const char* hb = (const char*)h1s + mrow * 512 + kt * 64 + hi * 16;
            a.q[0] = *(const u32x4*)hb;
            a.q[1] = *(const u32x4*)(hb + 32);
#pragma unroll
            for (int g = 0; g < 4; ++g)
#pragma unroll
                for (int p = 0; p < 2; ++p) {
                    int nt = g * 16 + 2 * wv + p;
                    const u32x4* bp = (const u32x4*)whh1 + (kt * 64 + nt) * 64 + lane * 2;
                    Frag b; b.q[0] = bp[0]; b.q[1] = bp[1];
                    acc[g][p] = __builtin_amdgcn_wmma_f32_16x16x32_bf16(
                        false, a.v, false, b.v, (short)0, acc[g][p], false, false);
                }
        }
        {   // x_t contribution: one zero-padded K=32 WMMA per gate tile
            Frag a;
#pragma unroll
            for (int j = 0; j < 16; ++j) {
                int k   = (j < 8 ? j : j + 8) + 8 * hi;
                float v = (k < Dn) ? x[xrow + t * Dn + k] : 0.f;
                a.s[j]  = f2bf(v);
            }
#pragma unroll
            for (int g = 0; g < 4; ++g)
#pragma unroll
                for (int p = 0; p < 2; ++p) {
                    int nt = g * 16 + 2 * wv + p;
                    const u32x4* bp = (const u32x4*)wih1 + nt * 64 + lane * 2;
                    Frag b; b.q[0] = bp[0]; b.q[1] = bp[1];
                    acc[g][p] = __builtin_amdgcn_wmma_f32_16x16x32_bf16(
                        false, a.v, false, b.v, (short)0, acc[g][p], false, false);
                }
        }
        __syncthreads();   // all waves done reading h1s(t-1)

#pragma unroll
        for (int p = 0; p < 2; ++p) {
            int col = 32 * wv + 16 * p + mrow;
#pragma unroll
            for (int r = 0; r < 8; ++r) {
                float ig = sigmoidf_(acc[0][p][r]);
                float fg = sigmoidf_(acc[1][p][r]);
                float gg = tanhf_(acc[2][p][r]);
                float og = sigmoidf_(acc[3][p][r]);
                float c  = fg * c1[p][r] + ig * gg;
                c1[p][r] = c;
                h1s[(r + 8 * hi) * Hn + col] = f2bf(og * tanhf_(c));
            }
        }
        __syncthreads();   // h1s(t) ready

        // ======================= layer 2 =======================
#pragma unroll
        for (int g = 0; g < 4; ++g)
#pragma unroll
            for (int p = 0; p < 2; ++p)
#pragma unroll
                for (int r = 0; r < 8; ++r) acc[g][p][r] = bs2[g][p];

#pragma unroll 2
        for (int kt = 0; kt < 8; ++kt) {
            Frag a1, a2;
            const char* hb1 = (const char*)h1s + mrow * 512 + kt * 64 + hi * 16;
            const char* hb2 = (const char*)h2s + mrow * 512 + kt * 64 + hi * 16;
            a1.q[0] = *(const u32x4*)hb1;  a1.q[1] = *(const u32x4*)(hb1 + 32);
            a2.q[0] = *(const u32x4*)hb2;  a2.q[1] = *(const u32x4*)(hb2 + 32);
#pragma unroll
            for (int g = 0; g < 4; ++g)
#pragma unroll
                for (int p = 0; p < 2; ++p) {
                    int nt = g * 16 + 2 * wv + p;
                    const u32x4* bp1 = (const u32x4*)wih2 + (kt * 64 + nt) * 64 + lane * 2;
                    const u32x4* bp2 = (const u32x4*)whh2 + (kt * 64 + nt) * 64 + lane * 2;
                    Frag b1; b1.q[0] = bp1[0]; b1.q[1] = bp1[1];
                    Frag b2; b2.q[0] = bp2[0]; b2.q[1] = bp2[1];
                    acc[g][p] = __builtin_amdgcn_wmma_f32_16x16x32_bf16(
                        false, a1.v, false, b1.v, (short)0, acc[g][p], false, false);
                    acc[g][p] = __builtin_amdgcn_wmma_f32_16x16x32_bf16(
                        false, a2.v, false, b2.v, (short)0, acc[g][p], false, false);
                }
        }
        __syncthreads();   // all waves done reading h2s(t-1)

#pragma unroll
        for (int p = 0; p < 2; ++p) {
            int col = 32 * wv + 16 * p + mrow;
#pragma unroll
            for (int r = 0; r < 8; ++r) {
                float ig = sigmoidf_(acc[0][p][r]);
                float fg = sigmoidf_(acc[1][p][r]);
                float gg = tanhf_(acc[2][p][r]);
                float og = sigmoidf_(acc[3][p][r]);
                float c  = fg * c2[p][r] + ig * gg;
                c2[p][r] = c;
                h2s[(r + 8 * hi) * Hn + col] = f2bf(og * tanhf_(c));
            }
        }
        __syncthreads();   // h2s(t) ready

        // =============== output projection: out_t = h2 @ Wout^T + bout ======
        {
            int m = tid >> 4, sseg = tid & 15;
            float partial[6] = {0.f, 0.f, 0.f, 0.f, 0.f, 0.f};
#pragma unroll
            for (int jj = 0; jj < 16; ++jj) {
                float hv = bf2f(h2s[m * Hn + sseg * 16 + jj]);
#pragma unroll
                for (int d = 0; d < Dn; ++d)
                    partial[d] += hv * Wout[d * Hn + sseg * 16 + jj];
            }
#pragma unroll
            for (int d = 0; d < Dn; ++d) pr[m][sseg][d] = partial[d];
        }
        __syncthreads();
        if (tid < 16 * Dn) {
            int m = tid / Dn, d = tid % Dn;
            float s = bout[d];
#pragma unroll
            for (int k2 = 0; k2 < 16; ++k2) s += pr[m][k2][d];
            out[(long)(bb + m) * (Tn * Dn) + t * Dn + d] = s;
        }
        __syncthreads();   // protect pr before next-step reuse
    }
}

// ---------------------------------------------------------------------------
extern "C" void kernel_launch(void* const* d_in, const int* in_sizes, int n_in,
                              void* d_out, int out_size, void* d_ws, size_t ws_size,
                              hipStream_t stream) {
    (void)in_sizes; (void)n_in; (void)out_size; (void)ws_size;
    const float* x    = (const float*)d_in[0];
    const float* Wih1 = (const float*)d_in[1];
    const float* Whh1 = (const float*)d_in[2];
    const float* bih1 = (const float*)d_in[3];
    const float* bhh1 = (const float*)d_in[4];
    const float* Wih2 = (const float*)d_in[5];
    const float* Whh2 = (const float*)d_in[6];
    const float* bih2 = (const float*)d_in[7];
    const float* bhh2 = (const float*)d_in[8];
    const float* Wout = (const float*)d_in[9];
    const float* bout = (const float*)d_in[10];

    unsigned short* wf = (unsigned short*)d_ws;
    float* biasb = (float*)((char*)d_ws + (size_t)2 * WF_TOTAL);

    const int total = WF_TOTAL + 2048;
    prep_weights<<<(total + 255) / 256, 256, 0, stream>>>(
        Wih1, Whh1, Wih2, Whh2, bih1, bhh1, bih2, bhh2, wf, biasb);

    lstm_persist<<<16, 256, 0, stream>>>(
        x, wf + WIH1_OFF, wf + WHH1_OFF, wf + WIH2_OFF, wf + WHH2_OFF,
        biasb, biasb + 1024, Wout, bout, (float*)d_out);
}
